// AmazonNet2_80547816669861
// MI455X (gfx1250) — matrix operand order repaired and checked by reading, not compiled, not run on hardware
//
#include <hip/hip_runtime.h>
#include <hip/hip_bf16.h>

// ---------------- problem constants (match reference) ----------------
#define NN    100000
#define ER    1600000
#define ET    (ER + NN)      // edges + self loops
#define NF    128
#define HID   64
#define HEADS 3
#define CHROM 21
#define D1    (HEADS * HID)  // 192
#define NG    8
#define NCLS  10

typedef __attribute__((ext_vector_type(16))) __bf16 v16bf;
typedef __attribute__((ext_vector_type(8)))  __bf16 v8bf;
typedef __attribute__((ext_vector_type(8)))  float  v8f;

// ---------------- small utility kernels ----------------
__global__ void k_fill_f32(float* __restrict__ p, float v, int n) {
  int t = blockIdx.x * blockDim.x + threadIdx.x;
  if (t < n) p[t] = v;
}

__global__ void k_f32_to_bf16(const float* __restrict__ s, __bf16* __restrict__ d, int n) {
  int t = blockIdx.x * blockDim.x + threadIdx.x;
  if (t < n) d[t] = (__bf16)s[t];
}

// transpose-convert weights: W [K, Nc] f32 row-major -> Wt [Ncpad, K] bf16 (pad cols with 0)
__global__ void k_wt(const float* __restrict__ W, __bf16* __restrict__ Wt,
                     int K, int Nc, int Ncpad, int total) {
  int t = blockIdx.x * blockDim.x + threadIdx.x;
  if (t >= total) return;                 // total = Ncpad * K
  int c = t / K, k = t - c * K;
  float v = (c < Nc) ? W[(size_t)k * Nc + c] : 0.f;
  Wt[(size_t)c * K + k] = (__bf16)v;
}

// ---------------- bf16 WMMA GEMM: Y[M,Ncol] = X[M,K] * W  (W given as Wt[Ncol,K]) ----
__global__ __launch_bounds__(256) void k_gemm_bf16(
    const __bf16* __restrict__ X, const __bf16* __restrict__ Wt,
    float* __restrict__ Y, int K, int Ncol, int nTiles) {
  int wid = (int)((blockIdx.x * (unsigned)blockDim.x + threadIdx.x) >> 5);
  if (wid >= nTiles) return;
  int lane = (int)(threadIdx.x & 31);
  int ntN = Ncol >> 4;
  int tM = wid / ntN, tN = wid - tM * ntN;
  int rc = lane & 15;
  int hw = lane >> 4;                     // half-wave select
  const __bf16* Abase = X  + (size_t)(tM * 16 + rc) * K + hw * 16;  // A: linear-K layout
  const __bf16* Bbase = Wt + (size_t)(tN * 16 + rc) * K + hw * 16;  // B: linear-K layout
  v8f acc = {0.f, 0.f, 0.f, 0.f, 0.f, 0.f, 0.f, 0.f};
  for (int k0 = 0; k0 < K; k0 += 32) {
    v8bf alo = *(const v8bf*)(Abase + k0);        // k = k0+hw*16 + 0..7
    v8bf ahi = *(const v8bf*)(Abase + k0 + 8);    // k = k0+hw*16 + 8..15
    v16bf a = __builtin_shufflevector(alo, ahi, 0,1,2,3,4,5,6,7,8,9,10,11,12,13,14,15);
    v16bf b = *(const v16bf*)(Bbase + k0);        // k = k0+hw*16 + 0..15
    acc = __builtin_amdgcn_wmma_f32_16x16x32_bf16(false, a, false, b,
                                                  (short)0, acc, false, false);
  }
  // D layout: lane -> col = tN*16+rc ; element i -> row = tM*16 + hw*8 + i
  float* Yp = Y + (size_t)(tM * 16 + hw * 8) * Ncol + (tN * 16 + rc);
  #pragma unroll
  for (int i = 0; i < 8; ++i) Yp[(size_t)i * Ncol] = acc[i];
}

// ---------------- edge pass A: logits + segment float-max ----------------
__device__ inline void atomicMaxF(float* addr, float v) {
  if (v >= 0.f) atomicMax((int*)addr, __float_as_int(v));
  else          atomicMin((unsigned int*)addr, __float_as_uint(v));
}

__global__ void k_edge_logits(const int* __restrict__ ei,
                              const float* __restrict__ xl, const float* __restrict__ xr,
                              const float* __restrict__ att,
                              float* __restrict__ logits, float* __restrict__ m,
                              int Eraw, int Etot, int ld, int H, int D) {
  int t = blockIdx.x * blockDim.x + threadIdx.x;
  if (t >= Etot * H) return;
  int e = t / H, h = t - e * H;
  int s, dn;
  if (e < Eraw) { s = ei[e]; dn = ei[Eraw + e]; }
  else          { s = e - Eraw; dn = s; }
  const float* pl = xl + (size_t)s  * ld + h * D;
  const float* pr = xr + (size_t)dn * ld + h * D;
  const float* pa = att + h * D;
  float acc = 0.f;
  for (int i = 0; i < D; ++i) {
    float v = pl[i] + pr[i];
    acc += (v > 0.f ? v : 0.2f * v) * pa[i];      // leaky_relu(0.2) . att
  }
  logits[t] = acc;
  atomicMaxF(m + (size_t)dn * H + h, acc);
}

// ---------------- edge pass B: exp, denom, weighted scatter ----------------
__global__ void k_edge_scatter(const int* __restrict__ ei, const float* __restrict__ xl,
                               const float* __restrict__ logits, const float* __restrict__ m,
                               float* __restrict__ z, float* __restrict__ agg,
                               int Eraw, int Etot, int ld, int H, int D) {
  int t = blockIdx.x * blockDim.x + threadIdx.x;
  if (t >= Etot * H) return;
  int e = t / H, h = t - e * H;
  int s, dn;
  if (e < Eraw) { s = ei[e]; dn = ei[Eraw + e]; }
  else          { s = e - Eraw; dn = s; }
  float ez = __expf(logits[t] - m[(size_t)dn * H + h]);
  atomicAdd(&z[(size_t)dn * H + h], ez);
  const float* pl = xl + (size_t)s * ld + h * D;
  float* po = agg + (size_t)dn * ld + h * D;
  for (int i = 0; i < D; ++i) atomicAdd(&po[i], ez * pl[i]);
}

// ---------------- layer-1 node pass: normalize + bias + relu ----------------
__global__ void k_node1(const float* __restrict__ agg, const float* __restrict__ z,
                        const float* __restrict__ bias, float* __restrict__ h, int total) {
  int t = blockIdx.x * blockDim.x + threadIdx.x;
  if (t >= total) return;
  int n = t / D1, j = t - n * D1;
  float v = agg[t] / z[n * HEADS + (j >> 6)] + bias[j];
  h[t] = v > 0.f ? v : 0.f;
}

// ---------------- batchnorm: stats then apply(+bf16 convert) ----------------
__global__ void k_bn_stats(const float* __restrict__ h, float* __restrict__ sums,
                           float* __restrict__ sumsq, int N) {
  int f = threadIdx.x;                              // blockDim.x == D1
  float s = 0.f, s2 = 0.f;
  for (int n = blockIdx.x; n < N; n += gridDim.x) {
    float v = h[(size_t)n * D1 + f];
    s += v; s2 += v * v;
  }
  atomicAdd(&sums[f], s);
  atomicAdd(&sumsq[f], s2);
}

__global__ void k_bn_apply(const float* __restrict__ h, const float* __restrict__ sums,
                           const float* __restrict__ sumsq, const float* __restrict__ gamma,
                           const float* __restrict__ beta, __bf16* __restrict__ out, int total) {
  int t = blockIdx.x * blockDim.x + threadIdx.x;
  if (t >= total) return;
  int f = t % D1;
  float mu  = sums[f] * (1.f / NN);
  float var = sumsq[f] * (1.f / NN) - mu * mu;
  float v = (h[t] - mu) * rsqrtf(var + 1e-5f) * gamma[f] + beta[f];
  out[t] = (__bf16)v;
}

// ---------------- head-mean node pass (layers 2 & 3) ----------------
__global__ void k_node_mean(const float* __restrict__ agg, const float* __restrict__ z,
                            const float* __restrict__ bias, float* __restrict__ res,
                            int N, int ld, int D) {
  int t = blockIdx.x * blockDim.x + threadIdx.x;
  if (t >= N * D) return;
  int n = t / D, d = t - n * D;
  float a = 0.f;
  #pragma unroll
  for (int h = 0; h < HEADS; ++h)
    a += agg[(size_t)n * ld + h * D + d] / z[n * HEADS + h];
  res[t] = a * (1.f / HEADS) + bias[d];
}

__global__ void k_relu_bf16(const float* __restrict__ s, __bf16* __restrict__ d, int n) {
  int t = blockIdx.x * blockDim.x + threadIdx.x;
  if (t < n) { float v = s[t]; d[t] = (__bf16)(v > 0.f ? v : 0.f); }
}

// ---------------- pooling + classifier ----------------
__global__ void k_pool(const float* __restrict__ h2, const int* __restrict__ batch,
                       float* __restrict__ pooled, float* __restrict__ cnt, int total) {
  int t = blockIdx.x * blockDim.x + threadIdx.x;
  if (t >= total) return;                           // total = NN*HID
  int n = t >> 6, d = t & 63;
  int g = batch[n];
  atomicAdd(&pooled[g * HID + d], h2[t]);
  if (d == 0) atomicAdd(&cnt[g], 1.f);
}

__global__ void k_classif(const float* __restrict__ pooled, const float* __restrict__ cnt,
                          const float* __restrict__ Wc, const float* __restrict__ bc,
                          float* __restrict__ out) {
  int t = blockIdx.x * blockDim.x + threadIdx.x;
  if (t >= NG * NCLS) return;
  int g = t / NCLS, c = t - g * NCLS;
  float inv = 1.f / fmaxf(cnt[g], 1.f);
  float a = 0.f;
  for (int d = 0; d < HID; ++d) a += pooled[g * HID + d] * inv * Wc[d * NCLS + c];
  out[t] = a + bc[c];
}

// ---------------- workspace layout ----------------
static constexpr size_t AL(size_t x) { return (x + 255) & ~size_t(255); }
static constexpr size_t SZ_A    = AL((size_t)NN * D1 * 2);     // bf16 activations / f32 h2
static constexpr size_t SZ_NODE = AL((size_t)NN * D1 * 4);     // xl / xr / agg
static constexpr size_t OFF_A  = 0;
static constexpr size_t OFF_B  = OFF_A + SZ_A;                 // xl / h1
static constexpr size_t OFF_C  = OFF_B + SZ_NODE;              // xr
static constexpr size_t OFF_D  = OFF_C + SZ_NODE;              // agg
static constexpr size_t OFF_E  = OFF_D + SZ_NODE;              // logits
static constexpr size_t SZ_E   = AL((size_t)ET * HEADS * 4);
static constexpr size_t OFF_M  = OFF_E + SZ_E;                 // segment max
static constexpr size_t SZ_MZ  = AL((size_t)NN * HEADS * 4);
static constexpr size_t OFF_Z  = OFF_M + SZ_MZ;                // segment denom
static constexpr size_t OFF_WL = OFF_Z + SZ_MZ;                // bf16 Wl^T
static constexpr size_t SZ_W   = AL((size_t)D1 * D1 * 2);
static constexpr size_t OFF_WR = OFF_WL + SZ_W;                // bf16 Wr^T
static constexpr size_t OFF_SM = OFF_WR + SZ_W;                // smalls
// smalls: pooled @+0 (2048B), cnt @+2048 (32B), bnsum @+2304 (768B), bnsq @+3072 (768B)

static inline int cdiv(long long a, long long b) { return (int)((a + b - 1) / b); }

extern "C" void kernel_launch(void* const* d_in, const int* in_sizes, int n_in,
                              void* d_out, int out_size, void* d_ws, size_t ws_size,
                              hipStream_t stream) {
  (void)in_sizes; (void)n_in; (void)out_size; (void)ws_size;
  const float* x     = (const float*)d_in[0];
  const int*   ei    = (const int*)  d_in[1];
  const int*   batch = (const int*)  d_in[2];
  const float* Wl1 = (const float*)d_in[3],  *Wr1 = (const float*)d_in[4];
  const float* att1= (const float*)d_in[5],  *b1  = (const float*)d_in[6];
  const float* gmm = (const float*)d_in[7],  *bet = (const float*)d_in[8];
  const float* Wl2 = (const float*)d_in[9],  *Wr2 = (const float*)d_in[10];
  const float* att2= (const float*)d_in[11], *b2  = (const float*)d_in[12];
  const float* Wl3 = (const float*)d_in[13], *Wr3 = (const float*)d_in[14];
  const float* att3= (const float*)d_in[15], *b3  = (const float*)d_in[16];
  const float* Wc  = (const float*)d_in[17], *bc  = (const float*)d_in[18];
  float* out = (float*)d_out;

  char* ws = (char*)d_ws;
  __bf16* A    = (__bf16*)(ws + OFF_A);
  float*  h2   = (float*) (ws + OFF_A);                         // reuses A after layer2 GEMM
  __bf16* h2b  = (__bf16*)(ws + OFF_A + (size_t)NN * HID * 4);  // relu(h2) bf16
  float*  XL   = (float*) (ws + OFF_B);
  float*  XR   = (float*) (ws + OFF_C);
  float*  AGG  = (float*) (ws + OFF_D);
  float*  LG   = (float*) (ws + OFF_E);
  float*  Mx   = (float*) (ws + OFF_M);
  float*  Zd   = (float*) (ws + OFF_Z);
  __bf16* WLt  = (__bf16*)(ws + OFF_WL);
  __bf16* WRt  = (__bf16*)(ws + OFF_WR);
  float*  pooled = (float*)(ws + OFF_SM);
  float*  cnt    = (float*)(ws + OFF_SM + 2048);
  float*  bnsum  = (float*)(ws + OFF_SM + 2304);
  float*  bnsq   = (float*)(ws + OFF_SM + 3072);

  const int B256 = 256;
  const int eth  = ET * HEADS;

  // ================= layer 1: 128 -> 3x64 concat =================
  k_wt<<<cdiv(D1 * NF, B256), B256, 0, stream>>>(Wl1, WLt, NF, D1, D1, D1 * NF);
  k_wt<<<cdiv(D1 * NF, B256), B256, 0, stream>>>(Wr1, WRt, NF, D1, D1, D1 * NF);
  k_f32_to_bf16<<<cdiv((long long)NN * NF, B256), B256, 0, stream>>>(x, A, NN * NF);
  {
    int nT = (NN / 16) * (D1 / 16);
    k_gemm_bf16<<<cdiv((long long)nT * 32, B256), B256, 0, stream>>>(A, WLt, XL, NF, D1, nT);
    k_gemm_bf16<<<cdiv((long long)nT * 32, B256), B256, 0, stream>>>(A, WRt, XR, NF, D1, nT);
  }
  k_fill_f32<<<cdiv(NN * HEADS, B256), B256, 0, stream>>>(Mx, -__builtin_inff(), NN * HEADS);
  k_fill_f32<<<cdiv(NN * HEADS, B256), B256, 0, stream>>>(Zd, 0.f, NN * HEADS);
  k_fill_f32<<<cdiv((long long)NN * D1, B256), B256, 0, stream>>>(AGG, 0.f, NN * D1);
  k_edge_logits<<<cdiv(eth, B256), B256, 0, stream>>>(ei, XL, XR, att1, LG, Mx, ER, ET, D1, HEADS, HID);
  k_edge_scatter<<<cdiv(eth, B256), B256, 0, stream>>>(ei, XL, LG, Mx, Zd, AGG, ER, ET, D1, HEADS, HID);
  k_node1<<<cdiv((long long)NN * D1, B256), B256, 0, stream>>>(AGG, Zd, b1, XL /*h1*/, NN * D1);
  // batchnorm
  k_fill_f32<<<1, D1, 0, stream>>>(bnsum, 0.f, D1);
  k_fill_f32<<<1, D1, 0, stream>>>(bnsq, 0.f, D1);
  k_bn_stats<<<512, D1, 0, stream>>>(XL, bnsum, bnsq, NN);
  k_bn_apply<<<cdiv((long long)NN * D1, B256), B256, 0, stream>>>(XL, bnsum, bnsq, gmm, bet, A, NN * D1);

  // ================= layer 2: 192 -> 3x64 mean =================
  k_wt<<<cdiv(D1 * D1, B256), B256, 0, stream>>>(Wl2, WLt, D1, D1, D1, D1 * D1);
  k_wt<<<cdiv(D1 * D1, B256), B256, 0, stream>>>(Wr2, WRt, D1, D1, D1, D1 * D1);
  {
    int nT = (NN / 16) * (D1 / 16);
    k_gemm_bf16<<<cdiv((long long)nT * 32, B256), B256, 0, stream>>>(A, WLt, XL, D1, D1, nT);
    k_gemm_bf16<<<cdiv((long long)nT * 32, B256), B256, 0, stream>>>(A, WRt, XR, D1, D1, nT);
  }
  k_fill_f32<<<cdiv(NN * HEADS, B256), B256, 0, stream>>>(Mx, -__builtin_inff(), NN * HEADS);
  k_fill_f32<<<cdiv(NN * HEADS, B256), B256, 0, stream>>>(Zd, 0.f, NN * HEADS);
  k_fill_f32<<<cdiv((long long)NN * D1, B256), B256, 0, stream>>>(AGG, 0.f, NN * D1);
  k_edge_logits<<<cdiv(eth, B256), B256, 0, stream>>>(ei, XL, XR, att2, LG, Mx, ER, ET, D1, HEADS, HID);
  k_edge_scatter<<<cdiv(eth, B256), B256, 0, stream>>>(ei, XL, LG, Mx, Zd, AGG, ER, ET, D1, HEADS, HID);
  k_node_mean<<<cdiv((long long)NN * HID, B256), B256, 0, stream>>>(AGG, Zd, b2, h2, NN, D1, HID);
  // pooling + classifier (uses pre-relu h2)
  k_fill_f32<<<1, B256, 0, stream>>>(pooled, 0.f, NG * HID);
  k_fill_f32<<<1, NG, 0, stream>>>(cnt, 0.f, NG);
  k_pool<<<cdiv((long long)NN * HID, B256), B256, 0, stream>>>(h2, batch, pooled, cnt, NN * HID);
  k_classif<<<1, 128, 0, stream>>>(pooled, cnt, Wc, bc, out);
  k_relu_bf16<<<cdiv((long long)NN * HID, B256), B256, 0, stream>>>(h2, h2b, NN * HID);

  // ================= layer 3: 64 -> 3x21 mean (pad N to 64) =================
  k_wt<<<cdiv(64 * HID, B256), B256, 0, stream>>>(Wl3, WLt, HID, HEADS * CHROM, 64, 64 * HID);
  k_wt<<<cdiv(64 * HID, B256), B256, 0, stream>>>(Wr3, WRt, HID, HEADS * CHROM, 64, 64 * HID);
  {
    int nT = (NN / 16) * (64 / 16);
    k_gemm_bf16<<<cdiv((long long)nT * 32, B256), B256, 0, stream>>>(h2b, WLt, XL, HID, 64, nT);
    k_gemm_bf16<<<cdiv((long long)nT * 32, B256), B256, 0, stream>>>(h2b, WRt, XR, HID, 64, nT);
  }
  k_fill_f32<<<cdiv(NN * HEADS, B256), B256, 0, stream>>>(Mx, -__builtin_inff(), NN * HEADS);
  k_fill_f32<<<cdiv(NN * HEADS, B256), B256, 0, stream>>>(Zd, 0.f, NN * HEADS);
  k_fill_f32<<<cdiv((long long)NN * 64, B256), B256, 0, stream>>>(AGG, 0.f, NN * 64);
  k_edge_logits<<<cdiv(eth, B256), B256, 0, stream>>>(ei, XL, XR, att3, LG, Mx, ER, ET, 64, HEADS, CHROM);
  k_edge_scatter<<<cdiv(eth, B256), B256, 0, stream>>>(ei, XL, LG, Mx, Zd, AGG, ER, ET, 64, HEADS, CHROM);
  k_node_mean<<<cdiv((long long)NN * CHROM, B256), B256, 0, stream>>>(AGG, Zd, b3, out + NG * NCLS, NN, 64, CHROM);
}
